// MultiHeadAttention_49941879717923
// MI455X (gfx1250) — matrix operand order repaired and checked by reading, not compiled, not run on hardware
//
#include <hip/hip_runtime.h>

// MI455X (gfx1250) MHA forward: bf16 WMMA (f32 accum) for all GEMMs,
// streamed softsign-attention (no softmax normalization -> no running stats).
// Double-buffered LDS pipeline in the GEMM, register-prefetched K/V chunks
// in the attention kernel. Fast-rcp softsign, fully unrolled register loops.

typedef __bf16 bf16;
typedef __attribute__((ext_vector_type(16))) __bf16 v16bf;
typedef __attribute__((ext_vector_type(8)))  float  v8f;

#define EMBED 1024
#define SEQ   2048
#define BATCH 4
#define HEADS 16
#define HDIM  64

__device__ __forceinline__ bf16 f2bf(float f) {
    unsigned u = __builtin_bit_cast(unsigned, f);
    u = (u + 0x7FFFu + ((u >> 16) & 1u)) >> 16;      // round-to-nearest-even
    unsigned short h = (unsigned short)u;
    return __builtin_bit_cast(bf16, h);
}

__device__ __forceinline__ v16bf make_v16(uint4 lo, uint4 hi) {
    union { uint4 u[2]; v16bf v; } x;
    x.u[0] = lo; x.u[1] = hi;
    return x.v;
}

// A-fragment (16x32 bf16, MxK): lane L holds row m = L&15, K-half = L>>4.
// Elements j=0..7 -> K = kh*8 + j ; j=8..15 -> K = 16 + kh*8 + (j-8).
__device__ __forceinline__ v16bf load_a_frag(const bf16* lds, int ldm,
                                             int mBase, int kBase, int lane) {
    int m  = mBase + (lane & 15);
    int kh = lane >> 4;
    const bf16* p = lds + m * ldm + kBase + kh * 8;
    uint4 lo = *(const uint4*)p;
    uint4 hi = *(const uint4*)(p + 16);
    return make_v16(lo, hi);
}

// B-fragment (32x16 bf16, KxN): lane L holds col n = L&15, K-half = L>>4.
// Elements j=0..7 -> K = kh*8 + j ; j=8..15 -> K = 16 + kh*8 + (j-8).
__device__ __forceinline__ v16bf load_b_frag(const bf16* lds, int ldn,
                                             int nBase, int kBase, int lane) {
    int n  = nBase + (lane & 15);
    int kh = lane >> 4;
    const bf16* p = lds + n * ldn + kBase + kh * 8;
    uint4 lo = *(const uint4*)p;
    uint4 hi = *(const uint4*)(p + 16);
    return make_v16(lo, hi);
}

// ---------------------------------------------------------------- cvt kernel
__global__ __launch_bounds__(256)
void cvt_f32_bf16(const float* __restrict__ in, bf16* __restrict__ out, int n) {
    int i = (blockIdx.x * blockDim.x + threadIdx.x) * 4;
    if (i < n) {
        float4 f = *(const float4*)(in + i);
        out[i + 0] = f2bf(f.x);
        out[i + 1] = f2bf(f.y);
        out[i + 2] = f2bf(f.z);
        out[i + 3] = f2bf(f.w);
    }
}

// --------------------------------------------------------- GEMM y = A*B^T + b
// A: [M,K] bf16 row-major, B: [N,K] bf16 row-major (torch Linear weight).
// MODE 0: bf16 out scattered to [B,H,S,D]; MODE 1: f32 out row-major [M,N].
// Double-buffered LDS: one barrier per K-step; global loads for step i+1
// are issued before the 8 WMMAs of step i.
#define BM 128
#define BN 128
#define BK 32
#define LDT 56   // padded tile stride (112B: 16B aligned, conflict-avoiding)

template <int MODE>
__global__ __launch_bounds__(256)
void gemm_bias_kernel(const bf16* __restrict__ A, const bf16* __restrict__ B,
                      const float* __restrict__ bias, void* __restrict__ out,
                      int M, int N, int K) {
    __shared__ bf16 At[2][BM * LDT];
    __shared__ bf16 Bt[2][BN * LDT];

    const int tid  = threadIdx.x;
    const int lane = tid & 31;
    const int wave = tid >> 5;     // 0..7
    const int mw   = wave & 1;     // 2 waves along M (64 rows each)
    const int nw   = wave >> 1;    // 4 waves along N (32 cols each)
    const int mBlock = blockIdx.x * BM;
    const int nBlock = blockIdx.y * BN;

    v8f c[4][2];
    const v8f vz = {0.f,0.f,0.f,0.f,0.f,0.f,0.f,0.f};
#pragma unroll
    for (int i = 0; i < 4; ++i)
#pragma unroll
        for (int j = 0; j < 2; ++j) c[i][j] = vz;

    const int rowHalf = tid >> 2;        // 0..63
    const int cg      = (tid & 3) * 8;   // 8-elem column group

    // prefetch tile 0 into registers, stage into buffer 0
    uint4 ra[2], rb[2];
#pragma unroll
    for (int rr = 0; rr < 2; ++rr) {
        int r = rowHalf + rr * 64;
        ra[rr] = *(const uint4*)(A + (size_t)(mBlock + r) * K + cg);
        rb[rr] = *(const uint4*)(B + (size_t)(nBlock + r) * K + cg);
    }
#pragma unroll
    for (int rr = 0; rr < 2; ++rr) {
        int r = rowHalf + rr * 64;
        *(uint4*)(&At[0][r * LDT + cg]) = ra[rr];
        *(uint4*)(&Bt[0][r * LDT + cg]) = rb[rr];
    }

    const int nIter = K / BK;
    for (int it = 0; it < nIter; ++it) {
        const int buf = it & 1;
        __syncthreads();   // buf stores visible; previous buf^1 reads retired

        // issue next tile's global loads (latency hides behind WMMAs below)
        if (it + 1 < nIter) {
            int k0 = (it + 1) * BK;
#pragma unroll
            for (int rr = 0; rr < 2; ++rr) {
                int r = rowHalf + rr * 64;
                ra[rr] = *(const uint4*)(A + (size_t)(mBlock + r) * K + k0 + cg);
                rb[rr] = *(const uint4*)(B + (size_t)(nBlock + r) * K + k0 + cg);
            }
        }

        v16bf a[4], b[2];
#pragma unroll
        for (int i = 0; i < 4; ++i) a[i] = load_a_frag(At[buf], LDT, mw * 64 + i * 16, 0, lane);
#pragma unroll
        for (int j = 0; j < 2; ++j) b[j] = load_b_frag(Bt[buf], LDT, nw * 32 + j * 16, 0, lane);
#pragma unroll
        for (int i = 0; i < 4; ++i)
#pragma unroll
            for (int j = 0; j < 2; ++j)
                c[i][j] = __builtin_amdgcn_wmma_f32_16x16x32_bf16(
                    false, a[i], false, b[j], (short)0, c[i][j], false, false);

        // stage next tile into the other buffer (reads of it were fenced by
        // the barrier at the top of this iteration)
        if (it + 1 < nIter) {
#pragma unroll
            for (int rr = 0; rr < 2; ++rr) {
                int r = rowHalf + rr * 64;
                *(uint4*)(&At[buf ^ 1][r * LDT + cg]) = ra[rr];
                *(uint4*)(&Bt[buf ^ 1][r * LDT + cg]) = rb[rr];
            }
        }
    }

    // Epilogue: C/D layout -> lane L: n = L&15; VGPR r: m = r + (L>>4)*8
#pragma unroll
    for (int i = 0; i < 4; ++i) {
#pragma unroll
        for (int j = 0; j < 2; ++j) {
            int n = nBlock + nw * 32 + j * 16 + (lane & 15);
            float bv = bias[n];
#pragma unroll
            for (int r = 0; r < 8; ++r) {
                int m = mBlock + mw * 64 + i * 16 + r + (lane >> 4) * 8;
                float val = c[i][j][r] + bv;
                if (MODE == 0) {
                    int bidx = m >> 11, s = m & 2047;
                    int h = n >> 6, d = n & 63;
                    ((bf16*)out)[(((size_t)(bidx * HEADS + h) * SEQ + s) << 6) + d] = f2bf(val);
                } else {
                    ((float*)out)[(size_t)m * N + n] = val;
                }
            }
        }
    }
}

// ------------------------------------------------------- streamed attention
// Per (b,h): ctx = softsign(Q K^T / 8) @ V.  Q,K,V bf16 [S,64] tiles.
// Block: 128 q-rows; stream key chunks of 64 with register prefetch of the
// next chunk overlapping both WMMA passes. LDS total 55.3 KB.
#define LQ 72
__global__ __launch_bounds__(256)
void attn_kernel(const bf16* __restrict__ Qg, const bf16* __restrict__ Kg,
                 const bf16* __restrict__ Vg, bf16* __restrict__ ctx) {
    __shared__ bf16 qT[128 * LQ];   // [qrow][e]
    __shared__ bf16 kT[64  * LQ];   // [krow][e]
    __shared__ bf16 vT[64  * LQ];   // transposed: [d][krow]
    __shared__ bf16 aT[128 * LQ];   // softsign(scores) bf16 [qrow][krow]

    const int bh = blockIdx.y;                 // 0..63
    const int m0 = blockIdx.x * 128;
    const bf16* q = Qg + (size_t)bh * SEQ * HDIM;
    const bf16* k = Kg + (size_t)bh * SEQ * HDIM;
    const bf16* v = Vg + (size_t)bh * SEQ * HDIM;

    const int tid  = threadIdx.x;
    const int lane = tid & 31;
    const int wave = tid >> 5;
    const int mw   = wave & 1;    // 64 q-rows per wave group (4 m-tiles)
    const int nw   = wave >> 1;   // 16-col tile (of 64) per wave

    const int lr = tid >> 3;        // 0..31 (chunk row group)
    const int lg = (tid & 7) * 8;   // 8-elem column group

    // load q tile [128x64]
#pragma unroll
    for (int rr = 0; rr < 4; ++rr) {
        int r = lr + rr * 32;
        *(uint4*)(qT + r * LQ + lg) = *(const uint4*)(q + (size_t)(m0 + r) * HDIM + lg);
    }

    // prefetch key/value chunk 0 into registers
    uint4 rk[2], rv[2];
#pragma unroll
    for (int rr = 0; rr < 2; ++rr) {
        int r = lr + rr * 32;
        rk[rr] = *(const uint4*)(k + (size_t)r * HDIM + lg);
        rv[rr] = *(const uint4*)(v + (size_t)r * HDIM + lg);
    }

    v8f c[4];
    const v8f vzero = {0.f,0.f,0.f,0.f,0.f,0.f,0.f,0.f};
#pragma unroll
    for (int i = 0; i < 4; ++i) c[i] = vzero;

    for (int kc = 0; kc < SEQ; kc += 64) {
        __syncthreads();   // prior kT/vT/aT reads retired; q tile visible
        // stage current chunk: kT row-major, vT transposed [d][krow]
#pragma unroll
        for (int rr = 0; rr < 2; ++rr) {
            int r = lr + rr * 32;
            *(uint4*)(kT + r * LQ + lg) = rk[rr];
            union { uint4 u; bf16 e[8]; } vv;
            vv.u = rv[rr];
#pragma unroll
            for (int i = 0; i < 8; ++i) vT[(lg + i) * LQ + r] = vv.e[i];
        }
        // issue next chunk's global loads (overlap with both WMMA passes)
        if (kc + 64 < SEQ) {
#pragma unroll
            for (int rr = 0; rr < 2; ++rr) {
                int r = lr + rr * 32;
                rk[rr] = *(const uint4*)(k + (size_t)(kc + 64 + r) * HDIM + lg);
                rv[rr] = *(const uint4*)(v + (size_t)(kc + 64 + r) * HDIM + lg);
            }
        }
        __syncthreads();

        // GEMM1: scores[128x64] over e=64
        v8f s[4];
#pragma unroll
        for (int i = 0; i < 4; ++i) s[i] = vzero;
#pragma unroll
        for (int ks = 0; ks < 64; ks += 32) {
            v16bf a[4];
#pragma unroll
            for (int i = 0; i < 4; ++i) a[i] = load_a_frag(qT, LQ, mw * 64 + i * 16, ks, lane);
            v16bf b = load_b_frag(kT, LQ, nw * 16, ks, lane);
#pragma unroll
            for (int i = 0; i < 4; ++i)
                s[i] = __builtin_amdgcn_wmma_f32_16x16x32_bf16(
                    false, a[i], false, b, (short)0, s[i], false, false);
        }
        // softsign(x/8) via fast v_rcp_f32 (bf16 output precision), -> aT
        {
            int n = nw * 16 + (lane & 15);
            int mB = mw * 64 + (lane >> 4) * 8;
#pragma unroll
            for (int i = 0; i < 4; ++i) {
#pragma unroll
                for (int r = 0; r < 8; ++r) {
                    float xv = s[i][r] * 0.125f;
                    float sv = xv * __builtin_amdgcn_rcpf(1.0f + fabsf(xv));
                    aT[(mB + i * 16 + r) * LQ + n] = f2bf(sv);
                }
            }
        }
        __syncthreads();

        // GEMM2: ctx += aT[128x64] @ V[64x64]  (B-frags from transposed vT)
#pragma unroll
        for (int ks = 0; ks < 64; ks += 32) {
            v16bf a[4];
#pragma unroll
            for (int i = 0; i < 4; ++i) a[i] = load_a_frag(aT, LQ, mw * 64 + i * 16, ks, lane);
            v16bf b = load_b_frag(vT, LQ, nw * 16, ks, lane);
#pragma unroll
            for (int i = 0; i < 4; ++i)
                c[i] = __builtin_amdgcn_wmma_f32_16x16x32_bf16(
                    false, a[i], false, b, (short)0, c[i], false, false);
        }
    }

    // write ctx -> [B,S,E] bf16 (merge heads)
    const int b_ = bh >> 4, h_ = bh & 15;
    const int d  = nw * 16 + (lane & 15);
#pragma unroll
    for (int i = 0; i < 4; ++i) {
#pragma unroll
        for (int r = 0; r < 8; ++r) {
            int mLoc = mw * 64 + i * 16 + r + (lane >> 4) * 8;
            ctx[(size_t)(b_ * SEQ + m0 + mLoc) * EMBED + h_ * HDIM + d] = f2bf(c[i][r]);
        }
    }
}

// ----------------------------------------------------------------- launcher
extern "C" void kernel_launch(void* const* d_in, const int* in_sizes, int n_in,
                              void* d_out, int out_size, void* d_ws, size_t ws_size,
                              hipStream_t stream) {
    const float* x  = (const float*)d_in[0];
    const float* Wq = (const float*)d_in[1];
    const float* bq = (const float*)d_in[2];
    const float* Wk = (const float*)d_in[3];
    const float* bk = (const float*)d_in[4];
    const float* Wv = (const float*)d_in[5];
    const float* bv = (const float*)d_in[6];
    const float* Wo = (const float*)d_in[7];
    const float* bo = (const float*)d_in[8];

    const int M = BATCH * SEQ;                 // 8192
    const size_t XN = (size_t)M * EMBED;       // 8,388,608
    const size_t WN = (size_t)EMBED * EMBED;   // 1,048,576

    char* ws = (char*)d_ws;
    size_t off = 0;
    auto take = [&](size_t bytes) { void* p = ws + off; off = (off + bytes + 255) & ~(size_t)255; return p; };

    bf16* xb   = (bf16*)take(XN * 2);
    bf16* wqb  = (bf16*)take(WN * 2);
    bf16* wkb  = (bf16*)take(WN * 2);
    bf16* wvb  = (bf16*)take(WN * 2);
    bf16* wob  = (bf16*)take(WN * 2);
    bf16* qb   = (bf16*)take(XN * 2);
    bf16* kb   = (bf16*)take(XN * 2);
    bf16* vb   = (bf16*)take(XN * 2);
    bf16* ctxb = (bf16*)take(XN * 2);

    // fp32 -> bf16 conversions
    cvt_f32_bf16<<<(int)(XN / 4 / 256), 256, 0, stream>>>(x,  xb,  (int)XN);
    cvt_f32_bf16<<<(int)(WN / 4 / 256), 256, 0, stream>>>(Wq, wqb, (int)WN);
    cvt_f32_bf16<<<(int)(WN / 4 / 256), 256, 0, stream>>>(Wk, wkb, (int)WN);
    cvt_f32_bf16<<<(int)(WN / 4 / 256), 256, 0, stream>>>(Wv, wvb, (int)WN);
    cvt_f32_bf16<<<(int)(WN / 4 / 256), 256, 0, stream>>>(Wo, wob, (int)WN);

    // Q/K/V projections (bias fused), scattered to [B,H,S,D] bf16
    dim3 gg(M / BM, EMBED / BN);
    gemm_bias_kernel<0><<<gg, 256, 0, stream>>>(xb, wqb, bq, qb, M, EMBED, EMBED);
    gemm_bias_kernel<0><<<gg, 256, 0, stream>>>(xb, wkb, bk, kb, M, EMBED, EMBED);
    gemm_bias_kernel<0><<<gg, 256, 0, stream>>>(xb, wvb, bv, vb, M, EMBED, EMBED);

    // streamed softsign attention
    dim3 ga(SEQ / 128, BATCH * HEADS);
    attn_kernel<<<ga, 256, 0, stream>>>(qb, kb, vb, ctxb);

    // output projection -> fp32 d_out
    gemm_bias_kernel<1><<<gg, 256, 0, stream>>>(ctxb, wob, bo, d_out, M, EMBED, EMBED);
}
